// FlaxMamba2Mixer_89687507076149
// MI455X (gfx1250) — matrix operand order repaired and checked by reading, not compiled, not run on hardware
//
#include <hip/hip_runtime.h>
#include <hip/hip_bf16.h>
#include <math.h>

// ---------------- problem constants ----------------
#define HIDN  2048
#define INTER 4096
#define NH    64
#define HD    64
#define NG    8
#define SS    128
#define CK    4
#define CHN   256          // chunk length
#define CONV_DIM 6144      // INTER + 2*NG*SS
#define PROJ  10304        // INTER + CONV_DIM + NH
#define NPAD1 10368        // PROJ padded to multiple of 128
#define B_SZ  2
#define SEQ   2048
#define BS    (B_SZ*SEQ)   // 4096 tokens
#define NCH   (SEQ/CHN)    // 8 chunks

typedef __attribute__((ext_vector_type(16))) __bf16 bf16x16;
typedef __attribute__((ext_vector_type(8)))  float  f32x8;

union FragU { unsigned int u[8]; bf16x16 v; };

__device__ __forceinline__ unsigned short f2bf(float x){
  unsigned int u = __float_as_uint(x);
  u += 0x7FFFu + ((u >> 16) & 1u);          // RNE
  return (unsigned short)(u >> 16);
}

// A-matrix fragment (16x32, bf16): lane half offsets K by 8 within each 16-block.
__device__ __forceinline__ bf16x16 load_fragA(const unsigned short* base, int ld, int r0, int k0){
  const int lane = threadIdx.x & 31;
  const int half = lane >> 4;
  const int r = lane & 15;
  const unsigned short* p = base + (r0 + r) * ld + k0 + half * 8;
  FragU f;
#pragma unroll
  for (int i = 0; i < 4; ++i){
    f.u[i]     = *(const unsigned int*)(p + 2*i);
    f.u[4 + i] = *(const unsigned int*)(p + 16 + 2*i);
  }
  return f.v;
}

// B-matrix fragment (32x16, bf16) from [N][K] LDS layout: lane half offsets K by 8 within each 16-block.
__device__ __forceinline__ bf16x16 load_fragB(const unsigned short* base, int ld, int n0, int k0){
  const int lane = threadIdx.x & 31;
  const int half = lane >> 4;
  const int n = lane & 15;
  const unsigned short* p = base + (n0 + n) * ld + k0 + half * 8;
  FragU f;
#pragma unroll
  for (int i = 0; i < 4; ++i){
    f.u[i]     = *(const unsigned int*)(p + 2*i);
    f.u[4 + i] = *(const unsigned int*)(p + 16 + 2*i);
  }
  return f.v;
}

__device__ __forceinline__ f32x8 wmma_bf16(bf16x16 a, bf16x16 b, f32x8 c){
  return __builtin_amdgcn_wmma_f32_16x16x32_bf16(false, a, false, b, (short)0, c, false, false);
}

__device__ __forceinline__ float silu_f(float x){ return x / (1.0f + __expf(-x)); }

// ---------------- packing kernels (f32 -> bf16, optional transpose) ----------------
__global__ __launch_bounds__(256) void k_pack_rows(
    const float* __restrict__ src, unsigned short* __restrict__ dst)
{
  size_t i = (size_t)blockIdx.x * 256 + threadIdx.x;
  dst[i] = f2bf(src[i]);
}

// src: [K][N] f32 row-major ; dst: [NP][K] bf16 (N-major, zero-padded rows)
__global__ __launch_bounds__(256) void k_pack_t(
    const float* __restrict__ src, unsigned short* __restrict__ dst, int K, int N)
{
  int n = blockIdx.y;
  int k = blockIdx.x * 256 + threadIdx.x;
  float v = (n < N) ? src[(size_t)k * N + n] : 0.0f;
  dst[(size_t)n * K + k] = f2bf(v);
}

// ---------------- 1/9: bf16 WMMA GEMM, async double-buffered LDS staging ----------------
#define TBM 128
#define TBN 128
#define TBK 64

__global__ __launch_bounds__(256) void k_gemm_async(
    const unsigned short* __restrict__ Ap,   // [M][K]  bf16
    const unsigned short* __restrict__ Bp,   // [NP][K] bf16 (N-major)
    float* __restrict__ O0, float* __restrict__ O1, float* __restrict__ O2,
    int K, int s1, int s2, int s3, int ld0, int ld1, int ld2)
{
  __shared__ unsigned short As[2][TBM][TBK];   // 2 x 16KB
  __shared__ unsigned short Bs[2][TBN][TBK];   // 2 x 16KB  (total 64KB)
  const int tid = threadIdx.x;
  const int m0 = blockIdx.y * TBM;
  const int n0 = blockIdx.x * TBN;
  const int w = tid >> 5;
  const int wm = (w & 3) * 32;
  const int wn = (w >> 2) * 64;
  f32x8 acc[2][4] = {};

  auto stage = [&](int buf, int k0){
    // A tile: 128 x 64 bf16 = 1024 x 16B chunks
#pragma unroll
    for (int i = 0; i < 4; ++i){
      int idx = tid + i * 256;
      int r = idx >> 3;
      int cc = (idx & 7) * 8;
      unsigned lds = (unsigned)(unsigned long long)&As[buf][r][cc];
      const unsigned short* g = Ap + (size_t)(m0 + r) * K + k0 + cc;
      asm volatile("global_load_async_to_lds_b128 %0, %1, off" :: "v"(lds), "v"(g) : "memory");
    }
    // B tile: 128 x 64 bf16
#pragma unroll
    for (int i = 0; i < 4; ++i){
      int idx = tid + i * 256;
      int r = idx >> 3;
      int cc = (idx & 7) * 8;
      unsigned lds = (unsigned)(unsigned long long)&Bs[buf][r][cc];
      const unsigned short* g = Bp + (size_t)(n0 + r) * K + k0 + cc;
      asm volatile("global_load_async_to_lds_b128 %0, %1, off" :: "v"(lds), "v"(g) : "memory");
    }
  };

  stage(0, 0);
  asm volatile("s_wait_asynccnt 0" ::: "memory");
  __syncthreads();
  int buf = 0;
  for (int k0 = 0; k0 < K; k0 += TBK){
    if (k0 + TBK < K) stage(buf ^ 1, k0 + TBK);   // prefetch next tile (async)
#pragma unroll
    for (int kk = 0; kk < 2; ++kk){
      bf16x16 a0 = load_fragA(&As[buf][0][0], TBK, wm,      kk*32);
      bf16x16 a1 = load_fragA(&As[buf][0][0], TBK, wm + 16, kk*32);
#pragma unroll
      for (int j = 0; j < 4; ++j){
        bf16x16 b = load_fragB(&Bs[buf][0][0], TBK, wn + j*16, kk*32);
        acc[0][j] = wmma_bf16(a0, b, acc[0][j]);
        acc[1][j] = wmma_bf16(a1, b, acc[1][j]);
      }
    }
    asm volatile("s_wait_asynccnt 0" ::: "memory");
    __syncthreads();
    buf ^= 1;
  }
  const int lane = tid & 31, half = lane >> 4, cn = lane & 15;
#pragma unroll
  for (int i = 0; i < 2; ++i)
#pragma unroll
    for (int j = 0; j < 4; ++j)
#pragma unroll
      for (int v = 0; v < 8; ++v){
        int gm = m0 + wm + i*16 + half*8 + v;
        int gn = n0 + wn + j*16 + cn;
        float val = acc[i][j][v];
        if (gn < s1)      O0[(size_t)gm*ld0 + gn] = val;
        else if (gn < s2) O1[(size_t)gm*ld1 + (gn - s1)] = val;
        else if (gn < s3) O2[(size_t)gm*ld2 + (gn - s2)] = val;
      }
}

// ---------------- 2: depthwise causal conv (SAME, k=4, lo=1) + SiLU ----------------
__global__ __launch_bounds__(256) void k_conv_silu(
    const float* __restrict__ xin, const float* __restrict__ cw,
    const float* __restrict__ cb, float* __restrict__ xout)
{
  int idx = blockIdx.x * 256 + threadIdx.x;
  int ch = idx % CONV_DIM;
  int rest = idx / CONV_DIM;
  int t = rest % SEQ;
  int b = rest / SEQ;
  float acc = cb[ch];
#pragma unroll
  for (int j = 0; j < CK; ++j){
    int tt = t - 1 + j;
    if (tt >= 0 && tt < SEQ)
      acc += xin[((size_t)(b*SEQ + tt))*CONV_DIM + ch] * cw[j*CONV_DIM + ch];
  }
  xout[(size_t)idx] = silu_f(acc);
}

// ---------------- 3: softplus(dt)+per-chunk cumsum of A*dt ----------------
__global__ __launch_bounds__(256) void k_dt_scan(
    const float* __restrict__ dt_raw, const float* __restrict__ dt_bias,
    const float* __restrict__ A_log, float* __restrict__ dt_soft,
    float* __restrict__ Acum, float* __restrict__ Alast)
{
  const int c = blockIdx.x, h = blockIdx.y, b = blockIdx.z;
  const int tid = threadIdx.x;
  const int t = c * CHN + tid;
  float x = dt_raw[((size_t)(b*SEQ + t))*NH + h] + dt_bias[h];
  float dtv = (x > 20.0f) ? x : log1pf(__expf(x));
  dtv = fmaxf(dtv, 0.0f);
  dt_soft[((size_t)(b*SEQ + t))*NH + h] = dtv;
  float abt = -__expf(A_log[h]) * dtv;
  __shared__ float sc[CHN];
  sc[tid] = abt;
  __syncthreads();
  for (int off = 1; off < CHN; off <<= 1){
    float v = (tid >= off) ? sc[tid - off] : 0.0f;
    __syncthreads();
    sc[tid] += v;
    __syncthreads();
  }
  Acum[(((size_t)(b*NCH + c))*NH + h)*CHN + tid] = sc[tid];
  if (tid == CHN-1) Alast[((size_t)(b*NH + h))*NCH + c] = sc[tid];
}

// ---------------- 4: intra-chunk Y_diag (+ D*hs) ----------------
__global__ __launch_bounds__(256) void k_ydiag(
    const float* __restrict__ xbc, const float* __restrict__ dts,
    const float* __restrict__ Acum, const float* __restrict__ Dv,
    float* __restrict__ y)
{
  const int lb = blockIdx.x;             // 0..3 (64-row l block)
  const int h  = blockIdx.y;
  const int bc = blockIdx.z;             // b*NCH + c
  const int b = bc >> 3, c = bc & 7;
  const int g = h >> 3;
  const int tid = threadIdx.x;
  const int w = tid >> 5;
  const int tr = w & 3;
  const int tc0 = (w >> 2) * 2;
  const int lane = tid & 31;
  const int half = lane >> 4;
  const int cn = lane & 15;

  __shared__ unsigned short Cl[64][128];   // [l][n]
  __shared__ unsigned short Bsh[64][128];  // [s][n]
  __shared__ unsigned short Ms[64][64];    // [l][s] decayed, masked
  __shared__ unsigned short Xst[64][64];   // [p][s]
  __shared__ float Acl[64];
  __shared__ float Acs[64];

  const int t0 = c * CHN;
  const size_t acbase = (((size_t)(b*NCH + c))*NH + h)*CHN;

  for (int i = tid; i < 64*128; i += 256){
    int lr = i >> 7, n = i & 127;
    int t = t0 + lb*64 + lr;
    Cl[lr][n] = f2bf(xbc[((size_t)(b*SEQ + t))*CONV_DIM + INTER + NG*SS + g*SS + n]);
  }
  if (tid < 64) Acl[tid] = Acum[acbase + lb*64 + tid];
  __syncthreads();

  f32x8 yacc[2] = {};
  for (int sb = 0; sb <= lb; ++sb){
    for (int i = tid; i < 64*128; i += 256){
      int sr = i >> 7, n = i & 127;
      int t = t0 + sb*64 + sr;
      Bsh[sr][n] = f2bf(xbc[((size_t)(b*SEQ + t))*CONV_DIM + INTER + g*SS + n]);
    }
    for (int i = tid; i < 64*64; i += 256){
      int sr = i >> 6, p = i & 63;
      int t = t0 + sb*64 + sr;
      float xv = xbc[((size_t)(b*SEQ + t))*CONV_DIM + h*HD + p] *
                 dts[((size_t)(b*SEQ + t))*NH + h];
      Xst[p][sr] = f2bf(xv);
    }
    if (tid < 64) Acs[tid] = Acum[acbase + sb*64 + tid];
    __syncthreads();
    // M = C * B^T over K = 128, then mask/decay
#pragma unroll
    for (int tcx = 0; tcx < 2; ++tcx){
      int tc = tc0 + tcx;
      f32x8 m = {};
#pragma unroll
      for (int kk = 0; kk < 4; ++kk){
        bf16x16 a  = load_fragA(&Cl[0][0], 128, tr*16, kk*32);
        bf16x16 bb = load_fragB(&Bsh[0][0], 128, tc*16, kk*32);
        m = wmma_bf16(a, bb, m);
      }
#pragma unroll
      for (int v = 0; v < 8; ++v){
        int ml = tr*16 + half*8 + v;
        int ns = tc*16 + cn;
        int gl = lb*64 + ml, gs = sb*64 + ns;
        float f = (gs <= gl) ? __expf(Acl[ml] - Acs[ns]) : 0.0f;
        Ms[ml][ns] = f2bf(m[v] * f);
      }
    }
    __syncthreads();
    // Y += M * X over K = 64
#pragma unroll
    for (int tcx = 0; tcx < 2; ++tcx){
      int tc = tc0 + tcx;
#pragma unroll
      for (int kk = 0; kk < 2; ++kk){
        bf16x16 a  = load_fragA(&Ms[0][0], 64, tr*16, kk*32);
        bf16x16 bb = load_fragB(&Xst[0][0], 64, tc*16, kk*32);
        yacc[tcx] = wmma_bf16(a, bb, yacc[tcx]);
      }
    }
    __syncthreads();
  }
  const float dscale = Dv[h];
#pragma unroll
  for (int tcx = 0; tcx < 2; ++tcx){
    int tc = tc0 + tcx;
#pragma unroll
    for (int v = 0; v < 8; ++v){
      int l = lb*64 + tr*16 + half*8 + v;
      int t = t0 + l;
      int p = tc*16 + cn;
      float hsv = xbc[((size_t)(b*SEQ + t))*CONV_DIM + h*HD + p];
      y[((size_t)(b*SEQ + t))*INTER + h*HD + p] = yacc[tcx][v] + dscale * hsv;
    }
  }
}

// ---------------- 5: per-chunk states = X^T @ (B * decay) ----------------
__global__ __launch_bounds__(256) void k_states(
    const float* __restrict__ xbc, const float* __restrict__ dts,
    const float* __restrict__ Acum, float* __restrict__ states)
{
  const int c = blockIdx.x, h = blockIdx.y, b = blockIdx.z;
  const int g = h >> 3;
  const int tid = threadIdx.x;
  const int w = tid >> 5;
  const int pr = (w & 3) * 16;
  const int nc0 = (w >> 2) * 4;
  __shared__ unsigned short Xt[64][256];    // [p][l]
  __shared__ unsigned short Bdt[128][64];   // [n][l-rel]
  __shared__ float Ac[CHN];
  const int t0 = c * CHN;
  const size_t acbase = (((size_t)(b*NCH + c))*NH + h)*CHN;
  Ac[tid] = Acum[acbase + tid];
  for (int i = tid; i < 64*256; i += 256){
    int l = i >> 6, p = i & 63;
    int t = t0 + l;
    float xv = xbc[((size_t)(b*SEQ + t))*CONV_DIM + h*HD + p] *
               dts[((size_t)(b*SEQ + t))*NH + h];
    Xt[p][l] = f2bf(xv);
  }
  __syncthreads();
  const float alast = Ac[CHN-1];
  f32x8 acc[4] = {};
  for (int lbk = 0; lbk < 4; ++lbk){
    for (int i = tid; i < 64*128; i += 256){
      int lr = i >> 7, n = i & 127;
      int t = t0 + lbk*64 + lr;
      float bv = xbc[((size_t)(b*SEQ + t))*CONV_DIM + INTER + g*SS + n];
      float dec = __expf(alast - Ac[lbk*64 + lr]);
      Bdt[n][lr] = f2bf(bv * dec);
    }
    __syncthreads();
#pragma unroll
    for (int ni = 0; ni < 4; ++ni){
#pragma unroll
      for (int kk = 0; kk < 2; ++kk){
        bf16x16 a  = load_fragA(&Xt[0][0], 256, pr, lbk*64 + kk*32);
        bf16x16 bb = load_fragB(&Bdt[0][0], 64, (nc0+ni)*16, kk*32);
        acc[ni] = wmma_bf16(a, bb, acc[ni]);
      }
    }
    __syncthreads();
  }
  const int lane = tid & 31, half = lane >> 4, cn = lane & 15;
  const size_t sbase = (((size_t)(b*NCH + c))*NH + h)*(size_t)(HD*SS);
#pragma unroll
  for (int ni = 0; ni < 4; ++ni)
#pragma unroll
    for (int v = 0; v < 8; ++v){
      int p = pr + half*8 + v;
      int n = (nc0+ni)*16 + cn;
      states[sbase + (size_t)p*SS + n] = acc[ni][v];
    }
}

// ---------------- 6: inter-chunk recurrence (in place) ----------------
__global__ __launch_bounds__(256) void k_chunkscan(
    const float* __restrict__ Alast, float* __restrict__ states)
{
  size_t idx = (size_t)blockIdx.x * 256 + threadIdx.x;  // B*NH*HD*SS
  int n = (int)(idx & (SS-1));
  int p = (int)((idx >> 7) & (HD-1));
  int h = (int)((idx >> 13) & (NH-1));
  int b = (int)(idx >> 19);
  float S = 0.0f;
  for (int c = 0; c < NCH; ++c){
    size_t sidx = ((((size_t)(b*NCH + c))*NH + h)*HD + p)*SS + n;
    float cur = states[sidx];
    states[sidx] = S;                     // state entering chunk c
    float a = Alast[((size_t)(b*NH + h))*NCH + c];
    S = cur + __expf(a) * S;
  }
}

// ---------------- 7: Y_off = (C @ states^T) * exp(Acum), added into y ----------------
__global__ __launch_bounds__(256) void k_yoff(
    const float* __restrict__ xbc, const float* __restrict__ Acum,
    const float* __restrict__ states, float* __restrict__ y)
{
  const int c = blockIdx.x, h = blockIdx.y, b = blockIdx.z;
  const int g = h >> 3;
  const int tid = threadIdx.x;
  const int w = tid >> 5;
  const int lane = tid & 31, half = lane >> 4, cn = lane & 15;
  __shared__ unsigned short Cb[128][128];  // [l-rel][n]
  __shared__ unsigned short St[64][128];   // [p][n]
  __shared__ float Ac[CHN];
  const int t0 = c * CHN;
  const size_t acbase = (((size_t)(b*NCH + c))*NH + h)*CHN;
  Ac[tid] = Acum[acbase + tid];
  const size_t sbase = (((size_t)(b*NCH + c))*NH + h)*(size_t)(HD*SS);
  for (int i = tid; i < 64*128; i += 256){
    int p = i >> 7, n = i & 127;
    St[p][n] = f2bf(states[sbase + (size_t)p*SS + n]);
  }
  for (int lbk = 0; lbk < 2; ++lbk){
    __syncthreads();
    for (int i = tid; i < 128*128; i += 256){
      int lr = i >> 7, n = i & 127;
      int t = t0 + lbk*128 + lr;
      Cb[lr][n] = f2bf(xbc[((size_t)(b*SEQ + t))*CONV_DIM + INTER + NG*SS + g*SS + n]);
    }
    __syncthreads();
    f32x8 acc[4] = {};
#pragma unroll
    for (int pi = 0; pi < 4; ++pi){
#pragma unroll
      for (int kk = 0; kk < 4; ++kk){
        bf16x16 a  = load_fragA(&Cb[0][0], 128, w*16, kk*32);
        bf16x16 bb = load_fragB(&St[0][0], 128, pi*16, kk*32);
        acc[pi] = wmma_bf16(a, bb, acc[pi]);
      }
    }
#pragma unroll
    for (int pi = 0; pi < 4; ++pi)
#pragma unroll
      for (int v = 0; v < 8; ++v){
        int l = lbk*128 + w*16 + half*8 + v;
        int t = t0 + l;
        int p = pi*16 + cn;
        y[((size_t)(b*SEQ + t))*INTER + h*HD + p] += acc[pi][v] * __expf(Ac[l]);
      }
  }
}

// ---------------- 8: gated RMSNorm, bf16 output (pre-packed A for out_proj) ----------------
__global__ __launch_bounds__(256) void k_gatenorm(
    const float* __restrict__ y, const float* __restrict__ gate,
    const float* __restrict__ norm_w, unsigned short* __restrict__ yn)
{
  const int tok = blockIdx.x;
  const int tid = threadIdx.x;
  float vals[16];
  float s = 0.0f;
#pragma unroll
  for (int i = 0; i < 16; ++i){
    int ch = tid + i*256;
    float gv = gate[(size_t)tok*INTER + ch];
    float f = y[(size_t)tok*INTER + ch] * silu_f(gv);
    vals[i] = f;
    s += f*f;
  }
  __shared__ float red[256];
  red[tid] = s;
  __syncthreads();
  for (int off = 128; off > 0; off >>= 1){
    if (tid < off) red[tid] += red[tid + off];
    __syncthreads();
  }
  float rs = rsqrtf(red[0] / (float)INTER + 1e-6f);
#pragma unroll
  for (int i = 0; i < 16; ++i){
    int ch = tid + i*256;
    yn[(size_t)tok*INTER + ch] = f2bf(norm_w[ch] * vals[i] * rs);
  }
}

// ---------------- launcher ----------------
extern "C" void kernel_launch(void* const* d_in, const int* in_sizes, int n_in,
                              void* d_out, int out_size, void* d_ws, size_t ws_size,
                              hipStream_t stream)
{
  (void)in_sizes; (void)n_in; (void)out_size; (void)ws_size;
  const float* x_in   = (const float*)d_in[0];
  const float* w_in   = (const float*)d_in[1];
  const float* conv_w = (const float*)d_in[2];
  const float* conv_b = (const float*)d_in[3];
  const float* dt_b   = (const float*)d_in[4];
  const float* A_log  = (const float*)d_in[5];
  const float* Dv     = (const float*)d_in[6];
  const float* nw     = (const float*)d_in[7];
  const float* w_out  = (const float*)d_in[8];
  float* out = (float*)d_out;

  char* ws = (char*)d_ws;
  size_t off = 0;
  auto allocf = [&](size_t nelem){ float* p = (float*)(ws + off); off += nelem * sizeof(float); return p; };
  auto alloch = [&](size_t nelem){ unsigned short* p = (unsigned short*)(ws + off);
                                   off += ((nelem * 2 + 15) & ~(size_t)15); return p; };
  float* gate    = allocf((size_t)BS * INTER);
  float* xbc_raw = allocf((size_t)BS * CONV_DIM);
  float* xbc_act = allocf((size_t)BS * CONV_DIM);
  float* dt_raw  = allocf((size_t)BS * NH);
  float* dt_soft = allocf((size_t)BS * NH);
  float* Acum    = allocf((size_t)B_SZ * NCH * NH * CHN);
  float* Alast   = allocf((size_t)B_SZ * NH * NCH);
  float* states  = allocf((size_t)B_SZ * NCH * NH * HD * SS);
  float* ybuf    = allocf((size_t)BS * INTER);
  unsigned short* Ax  = alloch((size_t)BS * HIDN);     // bf16 input tokens
  unsigned short* Bp1 = alloch((size_t)NPAD1 * HIDN);  // bf16 in_proj_w, N-major, padded
  unsigned short* Bp2 = alloch((size_t)HIDN * INTER);  // bf16 out_proj_w, N-major
  unsigned short* ynb = alloch((size_t)BS * INTER);    // bf16 normalized activations

  // 0. pre-pack GEMM operands to bf16 (B matrices N-major for contiguous tiles)
  k_pack_rows<<<(BS*HIDN)/256, 256, 0, stream>>>(x_in, Ax);
  k_pack_t<<<dim3(HIDN/256, NPAD1), 256, 0, stream>>>(w_in, Bp1, HIDN, PROJ);
  k_pack_t<<<dim3(INTER/256, HIDN), 256, 0, stream>>>(w_out, Bp2, INTER, HIDN);

  // 1. in_proj GEMM with split epilogue: gate | xBC | dt
  k_gemm_async<<<dim3(NPAD1/TBN, BS/TBM), 256, 0, stream>>>(
      Ax, Bp1, gate, xbc_raw, dt_raw,
      HIDN, INTER, INTER + CONV_DIM, PROJ, INTER, CONV_DIM, NH);
  // 2. depthwise conv + SiLU
  k_conv_silu<<<(BS*CONV_DIM)/256, 256, 0, stream>>>(xbc_raw, conv_w, conv_b, xbc_act);
  // 3. dt softplus + per-chunk cumsum of A*dt
  k_dt_scan<<<dim3(NCH, NH, B_SZ), 256, 0, stream>>>(dt_raw, dt_b, A_log, dt_soft, Acum, Alast);
  // 4. intra-chunk Y_diag + D residual
  k_ydiag<<<dim3(4, NH, B_SZ*NCH), 256, 0, stream>>>(xbc_act, dt_soft, Acum, Dv, ybuf);
  // 5. per-chunk states
  k_states<<<dim3(NCH, NH, B_SZ), 256, 0, stream>>>(xbc_act, dt_soft, Acum, states);
  // 6. inter-chunk recurrence (in place)
  k_chunkscan<<<(B_SZ*NH*HD*SS)/256, 256, 0, stream>>>(Alast, states);
  // 7. Y_off contribution
  k_yoff<<<dim3(NCH, NH, B_SZ), 256, 0, stream>>>(xbc_act, Acum, states, ybuf);
  // 8. gated RMSNorm -> bf16
  k_gatenorm<<<BS, 256, 0, stream>>>(ybuf, gate, nw, ynb);
  // 9. out_proj GEMM
  k_gemm_async<<<dim3(HIDN/TBN, BS/TBM), 256, 0, stream>>>(
      ynb, Bp2, out, out, out, INTER, HIDN, HIDN, HIDN, HIDN, HIDN, HIDN);
}
